// HNLQLinear_27015344292268
// MI455X (gfx1250) — matrix operand order repaired and checked by reading, not compiled, not run on hardware
//
#include <hip/hip_runtime.h>

// ---------- types ----------
typedef __attribute__((ext_vector_type(16))) __bf16 v16bf;
typedef __attribute__((ext_vector_type(8)))  float  v8f;
typedef unsigned int  uint32x4 __attribute__((ext_vector_type(4)));
typedef int           int32x8  __attribute__((ext_vector_type(8)));
typedef int           int32x4  __attribute__((ext_vector_type(4)));

union ABFrag { uint4 q[2]; v16bf v; };
union BF8Pack { unsigned short s[8]; uint4 v; };

#define Q_LATTICE 4
#define QMAX_ACT 127.0f

// RNE float -> bf16 bit pack
__device__ __forceinline__ unsigned short f2bf(float f) {
    unsigned u = __float_as_uint(f);
    u = u + 0x7FFFu + ((u >> 16) & 1u);
    return (unsigned short)(u >> 16);
}

// generic->LDS byte offset (LDS aperture base has zero low 32 bits)
__device__ __forceinline__ unsigned lds_off(const void* p) {
    return (unsigned)(unsigned long long)p;
}

// ---------- kernel 1: hierarchical nested-lattice weight quantization ----------
__global__ void k_quant_w(const float* __restrict__ W,
                          const float* __restrict__ theta, int theta_k,
                          const float* __restrict__ G,
                          const float* __restrict__ Gi,
                          unsigned short* __restrict__ Wq,
                          int O, int I) {
    __shared__ float sG[64], sGi[64];
    int t = threadIdx.x;
    if (t < 64) { sG[t] = G[t]; sGi[t] = Gi[t]; }
    __syncthreads();

    long gid = (long)blockIdx.x * blockDim.x + t;
    long nblk = (long)O * (I >> 3);
    if (gid >= nblk) return;

    int o = (int)(gid / (I >> 3));
    long base = gid << 3;

    float sc = 0.f;
    for (int k = 0; k < theta_k; ++k) sc += theta[(size_t)o * theta_k + k];
    sc /= (float)theta_k;

    const float4* wp = (const float4*)(W + base);
    float4 w0 = wp[0], w1 = wp[1];
    float w[8] = {w0.x, w0.y, w0.z, w0.w, w1.x, w1.y, w1.z, w1.w};
#pragma unroll
    for (int i = 0; i < 8; ++i) w[i] *= sc;

    float z[8];
#pragma unroll
    for (int j = 0; j < 8; ++j) {
        float s = 0.f;
#pragma unroll
        for (int i = 0; i < 8; ++i) s += w[i] * sGi[i * 8 + j];
        float zr = rintf(s);
        float m = zr + (float)(Q_LATTICE / 2);
        m -= (float)Q_LATTICE * floorf(m * (1.0f / (float)Q_LATTICE));
        z[j] = m - (float)(Q_LATTICE / 2);
    }

    BF8Pack out;
#pragma unroll
    for (int c = 0; c < 8; ++c) {
        float s = 0.f;
#pragma unroll
        for (int j = 0; j < 8; ++j) s += z[j] * sG[j * 8 + c];
        out.s[c] = f2bf(s);
    }
    *(uint4*)(Wq + base) = out.v;
}

// ---------- kernel 2: x fp32 -> bf16 ----------
__global__ void k_cvt_x(const float* __restrict__ x,
                        unsigned short* __restrict__ xb, long n8) {
    long gid = (long)blockIdx.x * blockDim.x + threadIdx.x;
    if (gid >= n8) return;
    const float4* xp = (const float4*)(x + (gid << 3));
    float4 a = xp[0], b = xp[1];
    BF8Pack out;
    out.s[0] = f2bf(a.x); out.s[1] = f2bf(a.y); out.s[2] = f2bf(a.z); out.s[3] = f2bf(a.w);
    out.s[4] = f2bf(b.x); out.s[5] = f2bf(b.y); out.s[6] = f2bf(b.z); out.s[7] = f2bf(b.w);
    *(uint4*)(xb + (gid << 3)) = out.v;
}

// ---------- kernel 3: WMMA GEMM (y = x @ Wq^T + bias) + fused LSQ ----------
// 256 threads = 8 waves; block tile 64(M) x 128(N); wave: 16 cols x 4 M-tiles.
// K-step = 64: TDM stages 64x64 bf16 A tile (128B row + 16B pad = 144B stride,
// bank-conflict free), double buffered; 8 WMMAs per barrier round.

#if defined(__gfx1250__) && __has_builtin(__builtin_amdgcn_tensor_load_to_lds) && __has_builtin(__builtin_amdgcn_s_wait_tensorcnt)
#define USE_TDM 1
#else
#define USE_TDM 0
#endif

__global__ void __launch_bounds__(256)
k_gemm(const unsigned short* __restrict__ xb,   // [M, K] bf16
       const unsigned short* __restrict__ wq,   // [N, K] bf16
       const float* __restrict__ bias,
       const float* __restrict__ alpha,
       float* __restrict__ out,                 // [M, N] f32
       int M, int N, int K) {
    const int lane = threadIdx.x & 31;
    const int wave = threadIdx.x >> 5;
    const int mBase = blockIdx.y * 64;
    const int nCol  = blockIdx.x * 128 + wave * 16 + (lane & 15);

    // double-buffered A tile: 64 rows x (128B data + 16B pad) = 9 x uint4 stride
    __shared__ uint4 sx[2][64][9];

    v8f acc0 = {}, acc1 = {}, acc2 = {}, acc3 = {};

    const int nsteps = K >> 6;  // K-steps of 64

#if USE_TDM
    // D#: data_size=2B (code 1), pad_enable, pad_interval=32 DW (code 4), pad_amount=4 DW (code 3)
    const int g1dw0 = (1 << 16) | (1 << 20) | (4 << 22) | (3 << 25);
    auto tdm_issue = [&](int kt, int b) {
        unsigned long long ga =
            (unsigned long long)(const void*)(xb + (size_t)mBase * K + kt);
        uint32x4 g0;
        g0[0] = 1u;                                    // count=1 valid descriptor
        g0[1] = lds_off(&sx[b][0][0]);                 // lds_addr
        g0[2] = (unsigned)ga;                          // global_addr[31:0]
        g0[3] = (unsigned)(ga >> 32) | (2u << 30);     // global_addr[56:32] | type=2
        int32x8 g1;
        g1[0] = g1dw0;
        g1[1] = (int)(((unsigned)K & 0xFFFFu) << 16);                         // tensor_dim0 lo
        g1[2] = (int)(((unsigned)K >> 16) | (((unsigned)M & 0xFFFFu) << 16)); // dim0 hi | dim1 lo
        g1[3] = (int)((((unsigned)M >> 16) & 0xFFFFu) | (64u << 16));         // dim1 hi | tile_dim0=64
        g1[4] = 64;                                    // tile_dim1=64, tile_dim2=0
        g1[5] = K;                                     // tensor_dim0_stride lo
        g1[6] = 0;
        g1[7] = 0;
        int32x4 z4 = {0, 0, 0, 0};
#if __has_include(<hip/amd_detail/amd_gfx1250_TDM.h>)
        int32x8 z8 = {0, 0, 0, 0, 0, 0, 0, 0};
        __builtin_amdgcn_tensor_load_to_lds(g0, g1, z4, z4, z8, 0);
#else
        __builtin_amdgcn_tensor_load_to_lds(g0, g1, z4, z4, 0);
#endif
    };
#else
    auto stage = [&](int kt, int b) {
        int row = threadIdx.x >> 2, ch = threadIdx.x & 3;
        const uint4* gp =
            (const uint4*)(xb + (size_t)(mBase + row) * K + kt + ch * 16);
        sx[b][row][ch * 2]     = gp[0];
        sx[b][row][ch * 2 + 1] = gp[1];
    };
#endif

    // prologue: stage k-step 0 into buffer 0
#if USE_TDM
    if (wave == 0) tdm_issue(0, 0);
#else
    stage(0, 0);
#endif

    const int bChunk = (lane < 16) ? 0 : 16;     // B: lanes 0-15 K=0..15, 16-31 K=16..31
    const int aSel   = (lane < 16) ? 0 : 2;      // A: first 16B chunk of this lane's 32B k-half

    int buf = 0;
    for (int s2 = 0; s2 < nsteps; ++s2) {
        const int kt = s2 << 6;
#if USE_TDM
        if (wave == 0) __builtin_amdgcn_s_wait_tensorcnt((short)0);
        __syncthreads();
        if (s2 + 1 < nsteps && wave == 0) tdm_issue(kt + 64, buf ^ 1);  // overlap next DMA
#else
        __syncthreads();
        if (s2 + 1 < nsteps) stage(kt + 64, buf ^ 1);
#endif
        // ---- both B fragments first (one global clause; latency overlaps LDS+WMMA) ----
        ABFrag bf0, bf1;
        {
            const uint4* bp = (const uint4*)(wq + (size_t)nCol * K + kt + bChunk);
            bf0.q[0] = bp[0];
            bf0.q[1] = bp[1];
            const uint4* bp1 = (const uint4*)(wq + (size_t)nCol * K + kt + 32 + bChunk);
            bf1.q[0] = bp1[0];
            bf1.q[1] = bp1[1];
        }

        // ---- k-chunk 0: batch all 8 ds_load_b128, then 4 back-to-back WMMAs ----
        {
            ABFrag a0, a1, a2, a3;
            const uint4* r0 = &sx[buf][ 0 + (lane & 15)][0];
            const uint4* r1 = &sx[buf][16 + (lane & 15)][0];
            const uint4* r2 = &sx[buf][32 + (lane & 15)][0];
            const uint4* r3 = &sx[buf][48 + (lane & 15)][0];
            a0.q[0] = r0[aSel]; a0.q[1] = r0[aSel + 1];
            a1.q[0] = r1[aSel]; a1.q[1] = r1[aSel + 1];
            a2.q[0] = r2[aSel]; a2.q[1] = r2[aSel + 1];
            a3.q[0] = r3[aSel]; a3.q[1] = r3[aSel + 1];
            acc0 = __builtin_amdgcn_wmma_f32_16x16x32_bf16(false, a0.v, false, bf0.v, (short)0, acc0, false, false);
            acc1 = __builtin_amdgcn_wmma_f32_16x16x32_bf16(false, a1.v, false, bf0.v, (short)0, acc1, false, false);
            acc2 = __builtin_amdgcn_wmma_f32_16x16x32_bf16(false, a2.v, false, bf0.v, (short)0, acc2, false, false);
            acc3 = __builtin_amdgcn_wmma_f32_16x16x32_bf16(false, a3.v, false, bf0.v, (short)0, acc3, false, false);
        }
        // ---- k-chunk 1 (uint4 indices 4..7 within the 144B row) ----
        {
            ABFrag a0, a1, a2, a3;
            const uint4* r0 = &sx[buf][ 0 + (lane & 15)][4];
            const uint4* r1 = &sx[buf][16 + (lane & 15)][4];
            const uint4* r2 = &sx[buf][32 + (lane & 15)][4];
            const uint4* r3 = &sx[buf][48 + (lane & 15)][4];
            a0.q[0] = r0[aSel]; a0.q[1] = r0[aSel + 1];
            a1.q[0] = r1[aSel]; a1.q[1] = r1[aSel + 1];
            a2.q[0] = r2[aSel]; a2.q[1] = r2[aSel + 1];
            a3.q[0] = r3[aSel]; a3.q[1] = r3[aSel + 1];
            acc0 = __builtin_amdgcn_wmma_f32_16x16x32_bf16(false, a0.v, false, bf1.v, (short)0, acc0, false, false);
            acc1 = __builtin_amdgcn_wmma_f32_16x16x32_bf16(false, a1.v, false, bf1.v, (short)0, acc1, false, false);
            acc2 = __builtin_amdgcn_wmma_f32_16x16x32_bf16(false, a2.v, false, bf1.v, (short)0, acc2, false, false);
            acc3 = __builtin_amdgcn_wmma_f32_16x16x32_bf16(false, a3.v, false, bf1.v, (short)0, acc3, false, false);
        }
        buf ^= 1;
    }

    // fused LSQ epilogue: a = relu(alpha)+1e-8; s = a/127; out = round(clip(y)/s)*s
    const float al  = alpha[0];
    const float aa  = fmaxf(al, 0.0f) + 1e-8f;
    const float sV  = aa * (1.0f / QMAX_ACT);
    const float inv = QMAX_ACT / aa;
    const float bv  = bias[nCol];
    const int rOff  = (lane < 16) ? 0 : 8;

#pragma unroll
    for (int mt = 0; mt < 4; ++mt) {
        v8f c = (mt == 0) ? acc0 : (mt == 1) ? acc1 : (mt == 2) ? acc2 : acc3;
#pragma unroll
        for (int v = 0; v < 8; ++v) {
            int row = mBase + mt * 16 + v + rOff;
            float y  = c[v] + bv;
            float yc = fminf(fmaxf(y, -aa), aa);
            float r  = rintf(yc * inv);       // RNE, matches jnp.round
            out[(size_t)row * N + nCol] = r * sV;
        }
    }
}

// ---------- launch ----------
extern "C" void kernel_launch(void* const* d_in, const int* in_sizes, int n_in,
                              void* d_out, int out_size, void* d_ws, size_t ws_size,
                              hipStream_t stream) {
    const float* x     = (const float*)d_in[0];
    const float* W     = (const float*)d_in[1];
    const float* bias  = (const float*)d_in[2];
    const float* theta = (const float*)d_in[3];
    const float* alpha = (const float*)d_in[4];
    const float* G     = (const float*)d_in[5];
    const float* Gi    = (const float*)d_in[6];

    const int O = in_sizes[2];              // 4096
    const int I = in_sizes[1] / O;          // 4096
    const int M = in_sizes[0] / I;          // 256
    const int theta_k = in_sizes[3] / O;    // 1

    unsigned short* xb  = (unsigned short*)d_ws;
    unsigned short* wqb = xb + (size_t)M * I;

    {   // x -> bf16
        long n8 = (long)M * I / 8;
        k_cvt_x<<<(int)((n8 + 255) / 256), 256, 0, stream>>>(x, xb, n8);
    }
    {   // weight lattice quantization -> bf16
        long nblk = (long)O * I / 8;
        k_quant_w<<<(int)((nblk + 255) / 256), 256, 0, stream>>>(W, theta, theta_k, G, Gi, wqb, O, I);
    }
    {   // WMMA GEMM + LSQ
        dim3 grid(O / 128, M / 64);
        k_gemm<<<grid, 256, 0, stream>>>(xb, wqb, bias, alpha, (float*)d_out, M, O, I);
    }
}
